// CNNFromScratch_68633577390333
// MI455X (gfx1250) — matrix operand order repaired and checked by reading, not compiled, not run on hardware
//
#include <hip/hip_runtime.h>
#include <hip/hip_bf16.h>
#include <stdint.h>

// ---------------------------------------------------------------------------
// CDNA5 / gfx1250 implementation of the CNN+MLP reference.
// All conv/MLP layers run through one templated WMMA GEMM:
//   - activations in (B, W, C) bf16 layout -> convs are GEMMs with contiguous
//     A rows (implicit im2col, row stride Cin)
//   - 64-column weight slab staged into LDS per workgroup with
//     global_load_async_to_lds_b128 (ASYNCcnt) + s_wait_asynccnt
//   - each wave32 owns a 16x64 tile: 4 x v_wmma_f32_16x16x32_bf16 accumulators
// ---------------------------------------------------------------------------

typedef __bf16 bf16_t;
typedef __bf16        v16bf __attribute__((ext_vector_type(16)));
typedef float         v8f   __attribute__((ext_vector_type(8)));
typedef unsigned int  u32x4 __attribute__((ext_vector_type(4)));

union FragBf {
    u32x4 h[2];   // two 16-byte halves loaded from memory/LDS
    v16bf v;      // reinterpret as the WMMA bf16 fragment
};

// ---------------------------------------------------------------------------
// Stage 0: transpose x (B, 512, 20) f32 -> xT (B, 20, 512) bf16 via LDS
// ---------------------------------------------------------------------------
__global__ __launch_bounds__(256) void transpose_x_kernel(
    const float* __restrict__ x, bf16_t* __restrict__ xT)
{
    __shared__ float tile[512 * 20];            // 40 KB
    const int b = blockIdx.x;
    const float* src = x + (size_t)b * (512 * 20);
    for (int i = threadIdx.x; i < 512 * 20; i += 256)
        tile[i] = src[i];                       // tile[c*20 + w], coalesced
    __syncthreads();
    bf16_t* dst = xT + (size_t)b * (20 * 512);
    for (int i = threadIdx.x; i < 20 * 512; i += 256) {
        int w = i >> 9;                         // i / 512
        int c = i & 511;                        // i % 512
        dst[i] = (bf16_t)tile[c * 20 + w];      // dst[w*512 + c]
    }
}

// ---------------------------------------------------------------------------
// Conv weight reorder: K[co, ci, kk] f32 -> W[co][kk*Cin + ci] bf16
// ---------------------------------------------------------------------------
__global__ void permute_conv_w_kernel(
    const float* __restrict__ in, bf16_t* __restrict__ out,
    int Cout, int Cin, int Kw)
{
    const int Ktot  = Cin * Kw;
    const int total = Cout * Ktot;
    int i = blockIdx.x * blockDim.x + threadIdx.x;
    if (i >= total) return;
    int co = i / Ktot;
    int r  = i - co * Ktot;
    int kk = r / Cin;
    int ci = r - kk * Cin;
    out[i] = (bf16_t)in[(size_t)co * Ktot + (size_t)ci * Kw + kk];
}

// MLP weight transpose: W[k, n] f32 (K x N) -> out[n*K + k] bf16
__global__ void transpose_mlp_w_kernel(
    const float* __restrict__ in, bf16_t* __restrict__ out, int K, int N)
{
    int i = blockIdx.x * blockDim.x + threadIdx.x;
    if (i >= K * N) return;
    int n = i / K;
    int k = i - n * K;
    out[i] = (bf16_t)in[(size_t)k * N + n];
}

// ---------------------------------------------------------------------------
// Templated bf16 WMMA GEMM with implicit-im2col rows and LDS-staged weights.
//   A row m (m = b*WOUT + wo) starts at A + (b*WIN + wo)*CIN, length KTOT.
//   Bw is stored [n*KTOT + k] (column-major, K contiguous).
//   Out[m*N + n] = max(0, sum) stored bf16.
// Block = 8 waves, all sharing one 64-column n-group; each wave owns a
// 16(M)x64(N) tile. Weight slab (64 x KTOT, +8 pad per column to kill LDS
// bank conflicts) is filled once with async global->LDS b128 copies.
// ---------------------------------------------------------------------------
template <int KTOT, int N, int WIN, int WOUT, int CIN>
__global__ __launch_bounds__(256) void gemm_bf16_wmma_lds(
    const bf16_t* __restrict__ A, const bf16_t* __restrict__ Bw,
    bf16_t* __restrict__ Out)
{
    constexpr int COLPAD  = KTOT + 8;          // bf16 elems per LDS column
    constexpr int NGROUPS = N / 64;
    __shared__ bf16_t Bs[64 * COLPAD];

    const int lane   = threadIdx.x & 31;
    const int wave   = threadIdx.x >> 5;
    const int ng     = (NGROUPS == 1) ? 0 : (int)(blockIdx.x % NGROUPS);
    const int mBlock = (NGROUPS == 1) ? (int)blockIdx.x : (int)(blockIdx.x / NGROUPS);
    const int n0     = ng * 64;

    // ---- async fill: 64 weight columns -> LDS (b128 chunks, ASYNCcnt) ----
    {
        constexpr int CHUNKS_PER_COL = KTOT / 8;        // 16B chunks per column
        constexpr int TOTAL = 64 * CHUNKS_PER_COL;      // exact multiple of 256
        const uint32_t ldsBase = (uint32_t)(uintptr_t)&Bs[0];
        const bf16_t* gBase = Bw + (size_t)n0 * KTOT;
        for (int i = threadIdx.x; i < TOTAL; i += 256) {
            int col = i / CHUNKS_PER_COL;
            int j   = i - col * CHUNKS_PER_COL;
            const bf16_t* g = gBase + (size_t)col * KTOT + (size_t)j * 8;
            uint32_t l = ldsBase + (uint32_t)(col * COLPAD + j * 8) * 2u;
            asm volatile("global_load_async_to_lds_b128 %0, %1, off"
                         :: "v"(l), "v"(g) : "memory");
        }
        asm volatile("s_wait_asynccnt 0x0" ::: "memory");
        __syncthreads();
    }

    const int hi = lane >> 4;                   // lane group 0 (0-15) / 1 (16-31)
    const int r  = lane & 15;

    // --- A row base (A-fragment: lane holds row M = lane&15) ---
    const int m  = (mBlock * 8 + wave) * 16 + r;
    const int b  = m / WOUT;
    const int wo = m - b * WOUT;
    const bf16_t* aRow = A + (size_t)(b * WIN + wo) * CIN;
    const int aSel = hi * 16;   // lanes 0-15: K 0..15; lanes 16-31: K 16..31

    // --- B column base in LDS (B-fragment: lane holds col N = lane&15) ---
    const bf16_t* bCol = Bs + (size_t)r * COLPAD;
    const int bSel = hi * 16;   // lanes 0-15: K 0..15; lanes 16-31: K 16..31

    v8f acc0 = {}, acc1 = {}, acc2 = {}, acc3 = {};

    for (int kb = 0; kb < KTOT; kb += 32) {
        FragBf fa;
        fa.h[0] = *(const u32x4*)(aRow + kb + aSel);
        fa.h[1] = *(const u32x4*)(aRow + kb + 8 + aSel);

        const bf16_t* bp = bCol + kb + bSel;
        FragBf fb0, fb1, fb2, fb3;
        fb0.h[0] = *(const u32x4*)(bp);
        fb0.h[1] = *(const u32x4*)(bp + 8);
        fb1.h[0] = *(const u32x4*)(bp + 16 * COLPAD);
        fb1.h[1] = *(const u32x4*)(bp + 16 * COLPAD + 8);
        fb2.h[0] = *(const u32x4*)(bp + 32 * COLPAD);
        fb2.h[1] = *(const u32x4*)(bp + 32 * COLPAD + 8);
        fb3.h[0] = *(const u32x4*)(bp + 48 * COLPAD);
        fb3.h[1] = *(const u32x4*)(bp + 48 * COLPAD + 8);

        acc0 = __builtin_amdgcn_wmma_f32_16x16x32_bf16(
            false, fa.v, false, fb0.v, (short)0, acc0, false, false);
        acc1 = __builtin_amdgcn_wmma_f32_16x16x32_bf16(
            false, fa.v, false, fb1.v, (short)0, acc1, false, false);
        acc2 = __builtin_amdgcn_wmma_f32_16x16x32_bf16(
            false, fa.v, false, fb2.v, (short)0, acc2, false, false);
        acc3 = __builtin_amdgcn_wmma_f32_16x16x32_bf16(
            false, fa.v, false, fb3.v, (short)0, acc3, false, false);
    }

    // --- Epilogue: C/D layout = VGPR g: (M = g + 8*hi, N = lane&15), ReLU ---
    const int mBase = (mBlock * 8 + wave) * 16 + hi * 8;
    const int nLane = n0 + r;
    v8f accs[4] = {acc0, acc1, acc2, acc3};
#pragma unroll
    for (int s = 0; s < 4; ++s) {
        const int n = nLane + s * 16;
#pragma unroll
        for (int g = 0; g < 8; ++g) {
            float v = fmaxf(accs[s][g], 0.0f);
            Out[(size_t)(mBase + g) * N + n] = (bf16_t)v;
        }
    }
}

// ---------------------------------------------------------------------------
// Max-pool (k=2,s=2) + reference flatten order:
//   out[b, c*4 + wp] = max(h3[b, 2wp, c], h3[b, 2wp+1, c]);  h3 is (B, 8, 256)
// ---------------------------------------------------------------------------
__global__ void pool_kernel(const bf16_t* __restrict__ h3,
                            bf16_t* __restrict__ hp, int Btot)
{
    int i = blockIdx.x * blockDim.x + threadIdx.x;
    if (i >= Btot * 1024) return;
    int b  = i >> 10;
    int rr = i & 1023;
    int c  = rr >> 2;
    int wp = rr & 3;
    const bf16_t* base = h3 + (size_t)b * (8 * 256);
    float v0 = (float)base[(2 * wp)     * 256 + c];
    float v1 = (float)base[(2 * wp + 1) * 256 + c];
    hp[i] = (bf16_t)fmaxf(v0, v1);
}

// Final layer: out[m] = dot(m2[m, 0:128], w3)   (no relu, f32 output)
__global__ void dot_out_kernel(const bf16_t* __restrict__ m2,
                               const float* __restrict__ w3,
                               float* __restrict__ out, int M, int K)
{
    int m = blockIdx.x * blockDim.x + threadIdx.x;
    if (m >= M) return;
    const bf16_t* row = m2 + (size_t)m * K;
    float s = 0.0f;
    for (int k = 0; k < K; ++k)
        s += (float)row[k] * w3[k];
    out[m] = s;
}

// ---------------------------------------------------------------------------
// Host-side orchestration
// ---------------------------------------------------------------------------
static inline size_t alignUp(size_t v, size_t a) { return (v + a - 1) & ~(a - 1); }

extern "C" void kernel_launch(void* const* d_in, const int* in_sizes, int n_in,
                              void* d_out, int out_size, void* d_ws, size_t ws_size,
                              hipStream_t stream) {
    (void)in_sizes; (void)n_in; (void)out_size; (void)ws_size;

    const float* x   = (const float*)d_in[0];   // (8192, 512, 20)
    const float* k1  = (const float*)d_in[1];   // (64, 512, 3)
    const float* k2  = (const float*)d_in[2];   // (128, 64, 5)
    const float* k3  = (const float*)d_in[3];   // (256, 128, 7)
    const float* mw1 = (const float*)d_in[4];   // (1024, 256)
    const float* mw2 = (const float*)d_in[5];   // (256, 128)
    const float* mw3 = (const float*)d_in[6];   // (128, 1)
    float* out = (float*)d_out;                 // (8192, 1)

    const int Bn = 8192;

    // Workspace carve-up (bf16 buffers, 256 B aligned)
    char* ws = (char*)d_ws;
    size_t off = 0;
    auto carve = [&](size_t elems) {
        size_t o = off;
        off = alignUp(off + elems * sizeof(bf16_t), 256);
        return (bf16_t*)(ws + o);
    };
    bf16_t* xT   = carve((size_t)Bn * 20 * 512);   // (B, 20, 512)
    bf16_t* w1b  = carve((size_t)64 * 1536);
    bf16_t* h1   = carve((size_t)Bn * 18 * 64);    // (B, 18, 64)
    bf16_t* w2b  = carve((size_t)128 * 320);
    bf16_t* h2   = carve((size_t)Bn * 14 * 128);   // (B, 14, 128)
    bf16_t* w3b  = carve((size_t)256 * 896);
    bf16_t* h3   = carve((size_t)Bn * 8 * 256);    // (B, 8, 256)
    bf16_t* hp   = carve((size_t)Bn * 1024);       // (B, 1024) pooled+flattened
    bf16_t* mw1b = carve((size_t)256 * 1024);
    bf16_t* m1   = carve((size_t)Bn * 256);
    bf16_t* mw2b = carve((size_t)128 * 256);
    bf16_t* m2   = carve((size_t)Bn * 128);

    // ---- Stage 0: layout transforms ----
    transpose_x_kernel<<<Bn, 256, 0, stream>>>(x, xT);

    permute_conv_w_kernel<<<(64 * 1536 + 255) / 256, 256, 0, stream>>>(k1, w1b, 64, 512, 3);
    permute_conv_w_kernel<<<(128 * 320 + 255) / 256, 256, 0, stream>>>(k2, w2b, 128, 64, 5);
    permute_conv_w_kernel<<<(256 * 896 + 255) / 256, 256, 0, stream>>>(k3, w3b, 256, 128, 7);
    transpose_mlp_w_kernel<<<(1024 * 256 + 255) / 256, 256, 0, stream>>>(mw1, mw1b, 1024, 256);
    transpose_mlp_w_kernel<<<(256 * 128 + 255) / 256, 256, 0, stream>>>(mw2, mw2b, 256, 128);

    // ---- Conv layers as WMMA GEMMs (grid = (M/128) * (N/64) blocks) ----
    // conv1: M = 8192*18 = 147456, N = 64, K = 1536
    gemm_bf16_wmma_lds<1536, 64, 20, 18, 512>
        <<<(Bn * 18 / 128) * 1, 256, 0, stream>>>(xT, w1b, h1);
    // conv2: M = 8192*14 = 114688, N = 128, K = 320
    gemm_bf16_wmma_lds<320, 128, 18, 14, 64>
        <<<(Bn * 14 / 128) * 2, 256, 0, stream>>>(h1, w2b, h2);
    // conv3: M = 8192*8 = 65536, N = 256, K = 896
    gemm_bf16_wmma_lds<896, 256, 14, 8, 128>
        <<<(Bn * 8 / 128) * 4, 256, 0, stream>>>(h2, w3b, h3);

    // ---- Max-pool + flatten (reference order: c*4 + wp) ----
    pool_kernel<<<(Bn * 1024 + 255) / 256, 256, 0, stream>>>(h3, hp, Bn);

    // ---- MLP layers ----
    gemm_bf16_wmma_lds<1024, 256, 1, 1, 1024>
        <<<(Bn / 128) * 4, 256, 0, stream>>>(hp, mw1b, m1);
    gemm_bf16_wmma_lds<256, 128, 1, 1, 256>
        <<<(Bn / 128) * 2, 256, 0, stream>>>(m1, mw2b, m2);

    // ---- Final N=1 layer (plain dot product, f32 out) ----
    dot_out_kernel<<<(Bn + 255) / 256, 256, 0, stream>>>(m2, mw3, out, Bn, 128);
}
